// CrossAttention_52072183497517
// MI455X (gfx1250) — matrix operand order repaired and checked by reading, not compiled, not run on hardware
//
#include <hip/hip_runtime.h>
#include <hip/hip_bf16.h>

typedef __bf16 bf16;
typedef __attribute__((ext_vector_type(16))) __bf16 v16bf;
typedef __attribute__((ext_vector_type(8)))  __bf16 v8bf;
typedef __attribute__((ext_vector_type(8)))  float  v8f;

#define DIM      1024
#define HEADS    16
#define HEAD_DIM 64
#define BATCH    4
#define NQ       4096
#define SCTX     512

// ---------------------------------------------------------------------------
// WMMA helper (CDNA5 wave32, 16x16x32 bf16 -> f32)
// ---------------------------------------------------------------------------
__device__ __forceinline__ v8f wmma_bf16(v16bf a, v16bf b, v8f c) {
  return __builtin_amdgcn_wmma_f32_16x16x32_bf16(false, a, false, b, (short)0, c,
                                                 false, false);
}

// A-matrix fragment (16x32 bf16, MxK). Lane L holds row M=L%16;
// element e has K = 16*(L>>4) + e  -> one 32B load.
__device__ __forceinline__ v16bf load_A_frag(const bf16* A, int ld, int m0, int k0) {
  int lane = threadIdx.x & 31;
  const bf16* p = A + (long)(m0 + (lane & 15)) * ld + k0 + ((lane >> 4) << 4);
  return *(const v16bf*)p;
}

// B-matrix fragment (32x16 bf16, KxN) from a row-major [N x K] buffer.
// Lane L holds column N=L%16, elements K = k0 + 16*(L>>4) + e -> one 32B load.
__device__ __forceinline__ v16bf load_B_frag(const bf16* Bt, int ld, int n0, int k0) {
  int lane = threadIdx.x & 31;
  const bf16* p = Bt + (long)(n0 + (lane & 15)) * ld + k0 + ((lane >> 4) << 4);
  return *(const v16bf*)p;
}

// ---------------------------------------------------------------------------
// DPP16 butterfly reductions within each 16-lane row (pure VALU, no LDS).
// Stages: xor1 (quad_perm[1,0,3,2]=0xB1), xor2 (quad_perm[2,3,0,1]=0x4E),
// row_half_mirror (0x141, partner quad), row_mirror (0x140, partner 8).
// ---------------------------------------------------------------------------
template <int CTRL>
__device__ __forceinline__ float dpp_mov_f32(float x) {
  return __int_as_float(
      __builtin_amdgcn_update_dpp(0, __float_as_int(x), CTRL, 0xF, 0xF, true));
}
__device__ __forceinline__ float row_max16(float x) {
  x = fmaxf(x, dpp_mov_f32<0xB1>(x));
  x = fmaxf(x, dpp_mov_f32<0x4E>(x));
  x = fmaxf(x, dpp_mov_f32<0x141>(x));
  x = fmaxf(x, dpp_mov_f32<0x140>(x));
  return x;
}
__device__ __forceinline__ float row_sum16(float x) {
  x += dpp_mov_f32<0xB1>(x);
  x += dpp_mov_f32<0x4E>(x);
  x += dpp_mov_f32<0x141>(x);
  x += dpp_mov_f32<0x140>(x);
  return x;
}

// ---------------------------------------------------------------------------
// fp32 -> bf16 convert (grid-stride)
// ---------------------------------------------------------------------------
__global__ void f32_to_bf16(const float* __restrict__ in, bf16* __restrict__ out, long n) {
  long i = (long)blockIdx.x * blockDim.x + threadIdx.x;
  long stride = (long)gridDim.x * blockDim.x;
  for (; i < n; i += stride) out[i] = (bf16)in[i];
}

// ---------------------------------------------------------------------------
// Weight transpose + convert: W[K][N] f32 -> Wt[N][K] bf16 (LDS tiled)
// grid = (N/32, K/32), block = (32, 8)
// ---------------------------------------------------------------------------
__global__ void transpose_w(const float* __restrict__ W, bf16* __restrict__ Wt,
                            int K, int N) {
  __shared__ float tile[32][33];
  int n0 = blockIdx.x * 32, k0 = blockIdx.y * 32;
  int tx = threadIdx.x, ty = threadIdx.y;
#pragma unroll
  for (int i = ty; i < 32; i += 8)
    tile[i][tx] = W[(long)(k0 + i) * N + n0 + tx];
  __syncthreads();
#pragma unroll
  for (int i = ty; i < 32; i += 8)
    Wt[(long)(n0 + i) * K + k0 + tx] = (bf16)tile[tx][i];
}

// ---------------------------------------------------------------------------
// Extract V from KV and transpose per head: KV[b][s][2*DIM] -> Vt[b*H+h][d][s]
// grid = (S/16, d/16, B*H), block = (16,16)
// ---------------------------------------------------------------------------
__global__ void v_transpose(const bf16* __restrict__ KV, bf16* __restrict__ Vt) {
  __shared__ bf16 t[16][17];
  int bh = blockIdx.z, b = bh >> 4, h = bh & 15;
  int s0 = blockIdx.x * 16, d0 = blockIdx.y * 16;
  int tx = threadIdx.x, ty = threadIdx.y;
  t[ty][tx] = KV[((long)b * SCTX + s0 + ty) * (2 * DIM) + DIM + h * HEAD_DIM + d0 + tx];
  __syncthreads();
  Vt[((long)bh * HEAD_DIM + d0 + ty) * SCTX + s0 + tx] = t[tx][ty];
}

// ---------------------------------------------------------------------------
// Generic bf16 GEMM: out[M][N] = A[M][K] @ Wt[N][K]^T + bias[N]
// One wave computes a 32x64 tile (2x4 WMMA accumulators, 8 WMMA / K-step).
// ---------------------------------------------------------------------------
template <bool OUT_F32>
__global__ void __launch_bounds__(256, 1)
gemm_bf16(const bf16* __restrict__ A, const bf16* __restrict__ Wt,
          const float* __restrict__ bias, void* __restrict__ out,
          int M, int N, int K) {
  int wid = (int)((blockIdx.x * blockDim.x + threadIdx.x) >> 5);
  int tn = N >> 6;                     // 64-wide N tiles
  int m0 = (wid / tn) << 5;
  int n0 = (wid % tn) << 6;
  int lane = threadIdx.x & 31;
  v8f acc[2][4] = {};
  for (int k0 = 0; k0 < K; k0 += 32) {
    if (k0 + 32 < K) {                 // prefetch next K-step tile rows
      __builtin_prefetch(A + (long)(m0 + lane) * K + k0 + 32, 0, 3);
      __builtin_prefetch(Wt + (long)(n0 + lane) * K + k0 + 32, 0, 3);
      __builtin_prefetch(Wt + (long)(n0 + 32 + lane) * K + k0 + 32, 0, 3);
    }
    v16bf a0 = load_A_frag(A, K, m0, k0);
    v16bf a1 = load_A_frag(A, K, m0 + 16, k0);
#pragma unroll
    for (int nt = 0; nt < 4; ++nt) {
      v16bf bfrg = load_B_frag(Wt, K, n0 + nt * 16, k0);
      acc[0][nt] = wmma_bf16(a0, bfrg, acc[0][nt]);
      acc[1][nt] = wmma_bf16(a1, bfrg, acc[1][nt]);
    }
  }
  int half = lane >> 4, nl = lane & 15;
#pragma unroll
  for (int nt = 0; nt < 4; ++nt) {
    float bv = bias[n0 + nt * 16 + nl];
#pragma unroll
    for (int mt = 0; mt < 2; ++mt)
#pragma unroll
      for (int r = 0; r < 8; ++r) {
        float v = acc[mt][nt][r] + bv;   // C layout: M = r + 8*half, N = nl
        long idx = (long)(m0 + mt * 16 + r + 8 * half) * N + n0 + nt * 16 + nl;
        if (OUT_F32) ((float*)out)[idx] = v;
        else         ((bf16*)out)[idx]  = (bf16)v;
      }
  }
}

// ---------------------------------------------------------------------------
// Flash attention: one wave per (b, h, 16-row Q tile). 64-column K/V chunks.
// Q[b*N+n][h*64+d] bf16, KV[b*S+s][2048] bf16 (K cols 0..1023),
// Vt[(b*16+h)*64+d][s] bf16, mask bool[b][s].  Output bf16 (b,n,h,d).
// __launch_bounds__(128,1): trade occupancy for zero spills (CDNA5 has up to
// 1024 VGPRs/wave; the chunk loop needs ~160 live VGPRs).
// ---------------------------------------------------------------------------
__global__ void __launch_bounds__(128, 1)
attn_fwd(const bf16* __restrict__ Q, const bf16* __restrict__ KV,
         const bf16* __restrict__ Vt,
         const unsigned char* __restrict__ mask,
         bf16* __restrict__ O) {
  __shared__ bf16 plds[4][16 * 64];           // per-wave P staging (16x64 row-major)
  int wslot = threadIdx.x >> 5;
  bf16* pw = &plds[wslot][0];
  int w  = blockIdx.x * 4 + wslot;
  int qt = w & 255;                           // N/16 = 256 q-tiles
  int h  = (w >> 8) & 15;
  int b  = w >> 12;
  int n0 = qt << 4;
  int lane = threadIdx.x & 31;
  int half = lane >> 4, nl = lane & 15;

  const bf16* Qb = Q + ((long)b * NQ + n0) * DIM + h * HEAD_DIM;
  v16bf qf0 = load_A_frag(Qb, DIM, 0, 0);     // d = 0..31
  v16bf qf1 = load_A_frag(Qb, DIM, 0, 32);    // d = 32..63

  float mrow[8], lrow[8];
#pragma unroll
  for (int r = 0; r < 8; ++r) { mrow[r] = -1e30f; lrow[r] = 0.f; }
  v8f oacc[4] = {};

  const unsigned char* mb = mask + b * SCTX;
  const bf16* Vb = Vt + (long)(b * HEADS + h) * HEAD_DIM * SCTX;

  for (int sc = 0; sc < SCTX; sc += 64) {
    if (sc + 64 < SCTX) {                     // prefetch next chunk K rows + V cols
      __builtin_prefetch(KV + ((long)b * SCTX + sc + 64 + lane) * (2 * DIM) + h * HEAD_DIM, 0, 3);
      __builtin_prefetch(KV + ((long)b * SCTX + sc + 96 + lane) * (2 * DIM) + h * HEAD_DIM, 0, 3);
      __builtin_prefetch(Vb + (long)lane * SCTX + sc + 64, 0, 3);
      __builtin_prefetch(Vb + (long)(lane + 32) * SCTX + sc + 64, 0, 3);
    }
    const bf16* Kb = KV + ((long)b * SCTX + sc) * (2 * DIM) + h * HEAD_DIM;
    v8f s[4];
#pragma unroll
    for (int t = 0; t < 4; ++t) {
      s[t] = v8f{};
      s[t] = wmma_bf16(qf0, load_B_frag(Kb, 2 * DIM, t * 16, 0),  s[t]);
      s[t] = wmma_bf16(qf1, load_B_frag(Kb, 2 * DIM, t * 16, 32), s[t]);
    }
    float msk[4];
#pragma unroll
    for (int t = 0; t < 4; ++t)
      msk[t] = mb[sc + t * 16 + nl] ? -1e30f : 0.f;

#pragma unroll
    for (int r = 0; r < 8; ++r) {            // row M = r + 8*half
      float v0 = s[0][r] * 0.125f + msk[0];
      float v1 = s[1][r] * 0.125f + msk[1];
      float v2 = s[2][r] * 0.125f + msk[2];
      float v3 = s[3][r] * 0.125f + msk[3];
      float cm = row_max16(fmaxf(fmaxf(v0, v1), fmaxf(v2, v3)));
      float mnew  = fmaxf(mrow[r], cm);
      float alpha = __expf(mrow[r] - mnew);
      float p0 = __expf(v0 - mnew);
      float p1 = __expf(v1 - mnew);
      float p2 = __expf(v2 - mnew);
      float p3 = __expf(v3 - mnew);
      float ps = row_sum16((p0 + p1) + (p2 + p3));
      lrow[r] = lrow[r] * alpha + ps;
      mrow[r] = mnew;
#pragma unroll
      for (int t = 0; t < 4; ++t) oacc[t][r] *= alpha;
      int row = r + 8 * half;                // stage P row-major 16x64 in LDS
      pw[row * 64 + nl]      = (bf16)p0;
      pw[row * 64 + 16 + nl] = (bf16)p1;
      pw[row * 64 + 32 + nl] = (bf16)p2;
      pw[row * 64 + 48 + nl] = (bf16)p3;
    }
    asm volatile("s_wait_dscnt 0" ::: "memory");   // P visible to A-layout reads
    v16bf pa0 = load_A_frag(pw, 64, 0, 0);         // P cols  0..31
    v16bf pa1 = load_A_frag(pw, 64, 0, 32);        // P cols 32..63
#pragma unroll
    for (int t = 0; t < 4; ++t) {
      oacc[t] = wmma_bf16(pa0, load_B_frag(Vb, SCTX, t * 16, sc),      oacc[t]);
      oacc[t] = wmma_bf16(pa1, load_B_frag(Vb, SCTX, t * 16, sc + 32), oacc[t]);
    }
  }

  bf16* Ob = O + ((long)b * NQ + n0) * DIM + h * HEAD_DIM;
#pragma unroll
  for (int r = 0; r < 8; ++r) {
    float inv = lrow[r] > 0.f ? 1.f / lrow[r] : 0.f;
#pragma unroll
    for (int t = 0; t < 4; ++t)
      Ob[(long)(r + 8 * half) * DIM + t * 16 + nl] = (bf16)(oacc[t][r] * inv);
  }
}

// ---------------------------------------------------------------------------
// Host launcher
// ---------------------------------------------------------------------------
extern "C" void kernel_launch(void* const* d_in, const int* in_sizes, int n_in,
                              void* d_out, int out_size, void* d_ws, size_t ws_size,
                              hipStream_t stream) {
  const float* x       = (const float*)d_in[0];
  const float* context = (const float*)d_in[1];
  const unsigned char* cmask = (const unsigned char*)d_in[2];  // jax bool = 1B
  const float* q_w    = (const float*)d_in[3];
  const float* q_b    = (const float*)d_in[4];
  const float* kv_w   = (const float*)d_in[5];
  const float* kv_b   = (const float*)d_in[6];
  const float* proj_w = (const float*)d_in[7];
  const float* proj_b = (const float*)d_in[8];
  float* out = (float*)d_out;

  char* ws = (char*)d_ws;
  size_t off = 0;
  auto alloc = [&](size_t bytes) -> char* {
    char* p = ws + off;
    off += (bytes + 255) & ~(size_t)255;
    return p;
  };
  bf16* xb   = (bf16*)alloc((size_t)BATCH * NQ * DIM * 2);          // 32 MB
  bf16* ctxb = (bf16*)alloc((size_t)BATCH * SCTX * DIM * 2);        //  4 MB
  bf16* qwt  = (bf16*)alloc((size_t)DIM * DIM * 2);                 //  2 MB
  bf16* kvwt = (bf16*)alloc((size_t)2 * DIM * DIM * 2);             //  4 MB
  bf16* pwt  = (bf16*)alloc((size_t)DIM * DIM * 2);                 //  2 MB
  bf16* Qb   = (bf16*)alloc((size_t)BATCH * NQ * DIM * 2);          // 32 MB
  bf16* KVb  = (bf16*)alloc((size_t)BATCH * SCTX * 2 * DIM * 2);    //  8 MB
  bf16* Vtb  = (bf16*)alloc((size_t)BATCH * HEADS * HEAD_DIM * SCTX * 2); // 4 MB
  bf16* attb = (bf16*)alloc((size_t)BATCH * NQ * DIM * 2);          // 32 MB

  f32_to_bf16<<<4096, 256, 0, stream>>>(x, xb, (long)BATCH * NQ * DIM);
  f32_to_bf16<<<2048, 256, 0, stream>>>(context, ctxb, (long)BATCH * SCTX * DIM);
  transpose_w<<<dim3(DIM / 32, DIM / 32), dim3(32, 8), 0, stream>>>(q_w, qwt, DIM, DIM);
  transpose_w<<<dim3(2 * DIM / 32, DIM / 32), dim3(32, 8), 0, stream>>>(kv_w, kvwt, DIM, 2 * DIM);
  transpose_w<<<dim3(DIM / 32, DIM / 32), dim3(32, 8), 0, stream>>>(proj_w, pwt, DIM, DIM);

  {  // Q = x @ q_w + q_b  (bf16 out)
    int M = BATCH * NQ, N = DIM;
    int waves = (M / 32) * (N / 64);
    gemm_bf16<false><<<waves / 8, 256, 0, stream>>>(xb, qwt, q_b, Qb, M, N, DIM);
  }
  {  // KV = context @ kv_w + kv_b  (bf16 out)
    int M = BATCH * SCTX, N = 2 * DIM;
    int waves = (M / 32) * (N / 64);
    gemm_bf16<false><<<waves / 8, 256, 0, stream>>>(ctxb, kvwt, kv_b, KVb, M, N, DIM);
  }
  v_transpose<<<dim3(SCTX / 16, HEAD_DIM / 16, BATCH * HEADS), dim3(16, 16), 0, stream>>>(KVb, Vtb);

  attn_fwd<<<(BATCH * HEADS * (NQ / 16)) / 4, 128, 0, stream>>>(Qb, KVb, Vtb, cmask, attb);

  {  // out = att @ proj_w + proj_b  (f32 out)
    int M = BATCH * NQ, N = DIM;
    int waves = (M / 32) * (N / 64);
    gemm_bf16<true><<<waves / 8, 256, 0, stream>>>(attb, pwt, proj_b, out, M, N, DIM);
  }
}
